// LinearAttention_49143015801154
// MI455X (gfx1250) — matrix operand order repaired and checked by reading, not compiled, not run on hardware
//
#include <hip/hip_runtime.h>
#include <hip/hip_bf16.h>

// ---------------------------------------------------------------------------
// LinearAttention decode step on MI455X (gfx1250, wave32, WMMA).
//   B=128, C=2048, H=16, D=128
// Roofline: kv_state 134MB read + 134MB write + 67MB weights ~= 340MB
// => ~15us floor at 23.3 TB/s. GEMMs (4.3 GFLOP) ride v_wmma_f32_16x16x32_bf16
// (f32 accumulate) so they stay far below the memory floor. f32->bf16 staging
// is done with single v_perm_b32 per packed pair to keep VALU off the
// critical path.
// ---------------------------------------------------------------------------

#define BATCH   128
#define EMBED   2048
#define HEADS   16
#define HDIM    128

typedef __attribute__((ext_vector_type(16))) __bf16 v16bf;
typedef __attribute__((ext_vector_type(8)))  float  v8f;

// Pack two f32 -> two bf16 (truncate) in ONE v_perm_b32:
// out[15:0] = lo[31:16], out[31:16] = hi[31:16]
__device__ __forceinline__ unsigned pack_bf16(unsigned lo, unsigned hi) {
  return __builtin_amdgcn_perm(hi, lo, 0x07060302u);
}

// ---------------------------------------------------------------------------
// GEMM body: C[128 x 2048] = act(A[128 x 2048] @ W[2048 x 2048] + bias)
// Block = 256 threads = 8 waves; block owns M=128 x N=64 (wave w: M rows
// [16w,16w+16), 4 accumulators over N columns [n0,n0+64)).
// Software-pipelined K loop: double-buffered LDS tiles (A 128x32 bf16,
// W transposed 64x32 bf16), float4 global prefetch of tile i+1 overlapped
// with WMMA on tile i; one barrier per iteration.
// ---------------------------------------------------------------------------
__device__ __forceinline__ void gemm_body(const float* __restrict__ A,
                                          const float* __restrict__ W,
                                          const float* __restrict__ bias,
                                          float* __restrict__ Cout,
                                          int act) {
  const int n0   = blockIdx.x * 64;
  const int tid  = threadIdx.x;
  const int wave = tid >> 5;
  const int lane = tid & 31;
  const int lh   = lane & 15;
  const bool hi  = lane >= 16;

  __shared__ __align__(16) unsigned short lds_a[2][BATCH * 32];  // [m][k] bf16
  __shared__ __align__(16) unsigned short lds_w[2][64 * 32];     // [n][k] bf16

  // A staging: thread owns 16 consecutive k at (m, kk) — 4x float4.
  const int a_base = tid * 16;          // linear m*32 + kk
  const int a_m    = a_base >> 5;
  const int a_kk   = a_base & 31;
  // W staging: thread owns a 2(k) x 4(n) patch — 2x float4 rows, packs pairs.
  const int w_n4   = (tid & 15) * 4;    // n offset within 64
  const int w_k2   = (tid >> 4) * 2;    // even k within 32

  float4 aIn[4];
  float4 wIn[2];

  auto load_tile = [&](int kb) {
    const float* asrc = A + (size_t)a_m * EMBED + kb + a_kk;        // 64B aligned
    aIn[0] = *(const float4*)(asrc + 0);
    aIn[1] = *(const float4*)(asrc + 4);
    aIn[2] = *(const float4*)(asrc + 8);
    aIn[3] = *(const float4*)(asrc + 12);
    const float* wsrc = W + (size_t)(kb + w_k2) * EMBED + n0 + w_n4; // 16B aligned
    wIn[0] = *(const float4*)(wsrc);          // row k2   : n4..n4+3
    wIn[1] = *(const float4*)(wsrc + EMBED);  // row k2+1 : n4..n4+3
  };
  auto store_tile = [&](int buf) {
    const unsigned* af = (const unsigned*)&aIn[0];
    unsigned* dstA = (unsigned*)&lds_a[buf][a_base];                 // 4B aligned
#pragma unroll
    for (int j = 0; j < 8; ++j)
      dstA[j] = pack_bf16(af[2 * j], af[2 * j + 1]);
    const unsigned* w0 = (const unsigned*)&wIn[0];
    const unsigned* w1 = (const unsigned*)&wIn[1];
#pragma unroll
    for (int j = 0; j < 4; ++j)   // [n][k] transpose: (k2,k2+1) pair at row n4+j
      *(unsigned*)&lds_w[buf][(w_n4 + j) * 32 + w_k2] = pack_bf16(w0[j], w1[j]);
  };

  v8f acc[4];
#pragma unroll
  for (int t = 0; t < 4; ++t) acc[t] = (v8f){0.f,0.f,0.f,0.f,0.f,0.f,0.f,0.f};

  load_tile(0);
  store_tile(0);
  __syncthreads();

  const int NT = EMBED / 32;   // 64 K-tiles
  for (int i = 0; i < NT; ++i) {
    const int buf = i & 1;
    if (i + 1 < NT) load_tile((i + 1) * 32);  // prefetch next tile (global)

    // A fragment (16x32 bf16): lanes 0-15 -> M=lh, K {0..7,16..23};
    //                          lanes 16-31 -> M=lh, K {8..15,24..31}
    union F { v16bf v; unsigned u[8]; };
    F af;
    {
      const unsigned short* row = &lds_a[buf][(wave * 16 + lh) * 32];
      const int kbase = hi ? 8 : 0;
#pragma unroll
      for (int r = 0; r < 4; ++r) af.u[r]     = *(const unsigned*)&row[kbase + 2 * r];
#pragma unroll
      for (int r = 0; r < 4; ++r) af.u[4 + r] = *(const unsigned*)&row[16 + kbase + 2 * r];
    }
    // All 4 B fragments first (32x16 bf16: lanes 0-15 -> N=lh, K {0..7,16..23};
    // lanes 16-31 -> K {8..15,24..31}), then WMMAs issue back-to-back.
    F bq[4];
    {
      const int kbase = hi ? 8 : 0;
#pragma unroll
      for (int t = 0; t < 4; ++t) {
        const unsigned short* col = &lds_w[buf][(t * 16 + lh) * 32];
#pragma unroll
        for (int r = 0; r < 8; ++r) bq[t].u[r] = *(const unsigned*)&col[kbase + 2 * r + (r >= 4 ? 8 : 0)];
      }
    }
#pragma unroll
    for (int t = 0; t < 4; ++t)
      acc[t] = __builtin_amdgcn_wmma_f32_16x16x32_bf16(
          false, af.v, false, bq[t].v, (short)0, acc[t], false, false);

    if (i + 1 < NT) store_tile((i + 1) & 1);  // fill the other buffer
    __syncthreads();
  }

  // Epilogue. C/D layout: VGPR r -> M = r (lanes 0-15) or r+8 (lanes 16-31), N = lh.
#pragma unroll
  for (int t = 0; t < 4; ++t) {
    const int n  = n0 + t * 16 + lh;
    const float bn = bias[n];
#pragma unroll
    for (int r = 0; r < 8; ++r) {
      const int m = wave * 16 + (hi ? r + 8 : r);
      float z = acc[t][r] + bn;
      if (act) {                                    // elu(z)+1, branchless
        const float e = __expf(fminf(z, 0.f));
        z = (z > 0.f) ? (z + 1.f) : e;
      }
      Cout[(size_t)m * EMBED + n] = z;
    }
  }
}

// grid = (32, 3): y selects Q/K/V so all three GEMMs fill the machine at once.
__global__ __launch_bounds__(256)
void qkv_gemm_kernel(const float* __restrict__ x,
                     const float* __restrict__ Wq, const float* __restrict__ bq,
                     const float* __restrict__ Wk, const float* __restrict__ bk,
                     const float* __restrict__ Wv, const float* __restrict__ bv,
                     float* __restrict__ qws, float* __restrict__ kws,
                     float* __restrict__ vws) {
  const float* W; const float* b; float* dst; int act;
  if (blockIdx.y == 0)      { W = Wq; b = bq; dst = qws; act = 1; }
  else if (blockIdx.y == 1) { W = Wk; b = bk; dst = kws; act = 1; }
  else                      { W = Wv; b = bv; dst = vws; act = 0; }
  gemm_body(x, W, b, dst, act);
}

__global__ __launch_bounds__(256)
void out_gemm_kernel(const float* __restrict__ attn,
                     const float* __restrict__ Wo, const float* __restrict__ bo,
                     float* __restrict__ out) {
  gemm_body(attn, Wo, bo, out, 0);
}

// ---------------------------------------------------------------------------
// Per-(b,h) state update — carries the 268MB kv_state read+write stream.
// Block = one head instance (128 threads): cx = column-quad (float4 columns
// 4cx..4cx+3), cy = row slice (d in [32cy, 32cy+32)). Fully-coalesced b128
// row streaming; den via LDS reduction; num partials reduced across cy.
// ---------------------------------------------------------------------------
__global__ __launch_bounds__(128)
void kv_update_kernel(const float* __restrict__ q, const float* __restrict__ k,
                      const float* __restrict__ v,
                      const float* __restrict__ kv_state,
                      const float* __restrict__ k_state,
                      float* __restrict__ kv_new, float* __restrict__ k_new,
                      float* __restrict__ attn) {
  const int bh  = blockIdx.x;            // b*16 + h
  const int tid = threadIdx.x;           // 0..127
  const int cx  = tid & 31;              // column quad
  const int cy  = tid >> 5;              // row slice
  const int bB  = bh >> 4, h = bh & 15;
  const size_t head_off = (size_t)bB * EMBED + h * HDIM;

  const float*  qv   = q + head_off;
  const float*  kv   = k + head_off;
  const float*  vv   = v + head_off;
  const float4* kvs4 = (const float4*)(kv_state + (size_t)bh * (HDIM * HDIM));
  float4*       kvo4 = (float4*)(kv_new + (size_t)bh * (HDIM * HDIM));

  __shared__ float  s_q[HDIM], s_k[HDIM], s_red[HDIM];
  __shared__ float4 s_num[4][32];

  const float qd = qv[tid];
  const float kd = kv[tid];
  s_q[tid] = qd;
  s_k[tid] = kd;
  const float kn = k_state[(size_t)bh * HDIM + tid] + kd;
  k_new[(size_t)bh * HDIM + tid] = kn;
  s_red[tid] = qd * kn;
  __syncthreads();
#pragma unroll
  for (int s = 64; s > 0; s >>= 1) {
    if (tid < s) s_red[tid] += s_red[tid + s];
    __syncthreads();
  }
  const float den = s_red[0] + 1e-6f;

  const float4 ve = *(const float4*)(vv + 4 * cx);
  float4 num = make_float4(0.f, 0.f, 0.f, 0.f);
  const int d0 = cy * 32;
#pragma unroll 4
  for (int d = d0; d < d0 + 32; ++d) {
    const float kdd = s_k[d];
    const float qdd = s_q[d];
    float4 t = kvs4[d * 32 + cx];
    t.x = fmaf(kdd, ve.x, t.x);
    t.y = fmaf(kdd, ve.y, t.y);
    t.z = fmaf(kdd, ve.z, t.z);
    t.w = fmaf(kdd, ve.w, t.w);
    kvo4[d * 32 + cx] = t;
    num.x = fmaf(qdd, t.x, num.x);
    num.y = fmaf(qdd, t.y, num.y);
    num.z = fmaf(qdd, t.z, num.z);
    num.w = fmaf(qdd, t.w, num.w);
  }
  s_num[cy][cx] = num;
  __syncthreads();
  if (cy == 0) {
    const float4 a = s_num[0][cx], b = s_num[1][cx],
                 c = s_num[2][cx], d = s_num[3][cx];
    float4 r;
    r.x = (a.x + b.x + c.x + d.x) / den;
    r.y = (a.y + b.y + c.y + d.y) / den;
    r.z = (a.z + b.z + c.z + d.z) / den;
    r.w = (a.w + b.w + c.w + d.w) / den;
    *(float4*)(attn + head_off + 4 * cx) = r;
  }
}

// ---------------------------------------------------------------------------
extern "C" void kernel_launch(void* const* d_in, const int* in_sizes, int n_in,
                              void* d_out, int out_size, void* d_ws, size_t ws_size,
                              hipStream_t stream) {
  (void)in_sizes; (void)n_in; (void)out_size; (void)ws_size;
  const float* x        = (const float*)d_in[0];
  const float* kv_state = (const float*)d_in[1];
  const float* k_state  = (const float*)d_in[2];
  const float* Wq = (const float*)d_in[3];
  const float* bq = (const float*)d_in[4];
  const float* Wk = (const float*)d_in[5];
  const float* bk = (const float*)d_in[6];
  const float* Wv = (const float*)d_in[7];
  const float* bv = (const float*)d_in[8];
  const float* Wo = (const float*)d_in[9];
  const float* bo = (const float*)d_in[10];

  float* out    = (float*)d_out;                                   // 128*2048
  float* kv_new = out + (size_t)BATCH * EMBED;                     // 128*16*128*128
  float* k_new  = kv_new + (size_t)BATCH * HEADS * HDIM * HDIM;    // 128*16*128

  float* q_ws = (float*)d_ws;
  float* k_ws = q_ws + (size_t)BATCH * EMBED;
  float* v_ws = k_ws + (size_t)BATCH * EMBED;
  float* a_ws = v_ws + (size_t)BATCH * EMBED;

  dim3 gqkv(EMBED / 64, 3);
  qkv_gemm_kernel<<<gqkv, 256, 0, stream>>>(x, Wq, bq, Wk, bk, Wv, bv,
                                            q_ws, k_ws, v_ws);
  kv_update_kernel<<<BATCH * HEADS, HDIM, 0, stream>>>(q_ws, k_ws, v_ws,
                                                       kv_state, k_state,
                                                       kv_new, k_new, a_ws);
  out_gemm_kernel<<<EMBED / 64, 256, 0, stream>>>(a_ws, Wo, bo, out);
}
